// lstm_32615981646159
// MI455X (gfx1250) — compile-verified, matrix operand order checked by reading, not run on hardware
//
#include <hip/hip_runtime.h>
#include <hip/hip_bf16.h>

// ---------------------------------------------------------------------------
// Fused LSTM-cell pipeline for MI455X (gfx1250, wave32, WMMA).
//   gates = input @ W_ih^T + (b_ih + b_hh)   [M=65536, N=200, K=1024]
//   h     = sigm(o) * tanh(sigm(i) * tanh(g))
//   out   = h @ W_lin^T + b_lin              [M=65536, N=1024, K=50]
// Memory-bound (512 MB HBM traffic @ 23.3 TB/s ~ 22us); everything fused so
// input is read once and out written once. bf16 WMMA, fp32 accumulate.
// ---------------------------------------------------------------------------

typedef __attribute__((ext_vector_type(16))) __bf16 v16bf;
typedef __attribute__((ext_vector_type(8)))  __bf16 v8bf;
typedef __attribute__((ext_vector_type(4)))  __bf16 v4bf;
typedef __attribute__((ext_vector_type(8)))  float  v8f;

union FragBF { v16bf f; v8bf h[2]; };

#define M_TOTAL 65536
#define F_DIM   1024
#define HID_SZ  50
#define NG      208          // 4*HID padded to 13 * 16
#define NT1     13
#define HPAD    64           // HID padded to 2 * 32
#define BLOCK_M 128
#define KC      64           // input K-chunk
#define NCHUNK  (F_DIM / KC) // 16
#define INS     72           // s_in row stride (bf16), conflict-free pad
#define GS      212          // gates row stride (f32)
#define HS      72           // s_h row stride (bf16)

// workspace layout (bytes)
#define WG_OFF   0
#define WL_OFF   (NG * F_DIM * 2)            // 425984
#define BIAS_OFF (WL_OFF + F_DIM * HPAD * 2) // 557056
// total ws use: 557888 bytes

__device__ __forceinline__ float sigm_f(float x) {
    return 1.0f / (1.0f + __expf(-x));
}
__device__ __forceinline__ float tanh_f(float x) {
    return 2.0f * sigm_f(2.0f * x) - 1.0f;
}

// ---- prep: pack weights to bf16 (padded) + fold biases --------------------
__global__ void lstm_prep_weights(const float* __restrict__ W_ih,
                                  const float* __restrict__ b_ih,
                                  const float* __restrict__ b_hh,
                                  const float* __restrict__ W_lin,
                                  __bf16* __restrict__ Wg,
                                  __bf16* __restrict__ Wl,
                                  float* __restrict__ biasC) {
    int idx = blockIdx.x * 256 + threadIdx.x;
    if (idx < NG * F_DIM) {
        int n = idx >> 10, k = idx & (F_DIM - 1);
        Wg[idx] = (n < 4 * HID_SZ) ? (__bf16)W_ih[n * F_DIM + k] : (__bf16)0.0f;
    }
    if (idx < F_DIM * HPAD) {
        int f = idx >> 6, k = idx & (HPAD - 1);
        Wl[idx] = (k < HID_SZ) ? (__bf16)W_lin[f * HID_SZ + k] : (__bf16)0.0f;
    }
    if (idx < NG) {
        biasC[idx] = (idx < 4 * HID_SZ) ? (b_ih[idx] + b_hh[idx]) : 0.0f;
    }
}

// ---- fused main kernel: one 128-row block per workgroup (8 waves) ---------
__global__ __launch_bounds__(256)
void lstm_fused(const float* __restrict__ in,
                const __bf16* __restrict__ Wg,
                const __bf16* __restrict__ Wl,
                const float* __restrict__ biasC,
                const float* __restrict__ b_lin,
                float* __restrict__ out) {
    // LDS: input staging (double buffered) aliases the gates buffer (phases
    // are barrier-separated); h buffer is separate. Total 124 KB -> 2 WG/WGP.
    __shared__ alignas(32) char smem[BLOCK_M * GS * 4 + BLOCK_M * HS * 2];
    __bf16 (*s_in)[BLOCK_M][INS] = (__bf16 (*)[BLOCK_M][INS])smem; // [2][128][72]
    float (*s_gates)[GS] = (float (*)[GS])smem;                     // [128][212]
    __bf16 (*s_h)[HS] = (__bf16 (*)[HS])(smem + BLOCK_M * GS * 4);  // [128][72]

    const int tid   = threadIdx.x;
    const int wid   = tid >> 5;      // wave id 0..7 -> owns m-tile wid
    const int lane  = tid & 31;
    const int lrow  = lane & 15;     // row (A) / col (B,C,D) within tile
    const int lhalf = lane >> 4;     // k-half for A/B frags, +8 rows for C/D
    const int rowBlock = blockIdx.x * BLOCK_M;

    // staging map: thread -> (row, 8 float4's of a 64-col chunk)
    const int sr = tid >> 1;
    const int sc = (tid & 1) * 8;

    const v8f vz = {0.f, 0.f, 0.f, 0.f, 0.f, 0.f, 0.f, 0.f};
    v8f acc[NT1];
#pragma unroll
    for (int i = 0; i < NT1; ++i) acc[i] = vz;

    const float4* in4 = (const float4*)in;
    float4 pref[8];

    // ---- load chunk 0 into buffer 0 ----
#pragma unroll
    for (int i = 0; i < 8; ++i)
        pref[i] = in4[(size_t)(rowBlock + sr) * (F_DIM / 4) + (sc + i)];
#pragma unroll
    for (int i = 0; i < 8; ++i) {
        v4bf b; b.x = (__bf16)pref[i].x; b.y = (__bf16)pref[i].y;
        b.z = (__bf16)pref[i].z; b.w = (__bf16)pref[i].w;
        *(v4bf*)&s_in[0][sr][(sc + i) * 4] = b;
    }
    __syncthreads();

    // ---- GEMM1 K loop: 16 chunks x 2 k-steps x 13 n-tiles of WMMA --------
    for (int c = 0; c < NCHUNK; ++c) {
        const int cur = c & 1;
        const bool havenext = (c + 1) < NCHUNK;
        if (havenext) {
#pragma unroll
            for (int i = 0; i < 8; ++i)
                pref[i] = in4[(size_t)(rowBlock + sr) * (F_DIM / 4) +
                              (c + 1) * (KC / 4) + (sc + i)];
        }
#pragma unroll
        for (int ks = 0; ks < 2; ++ks) {
            const int kloc = ks * 32 + lhalf * 8;
            FragBF a;
            a.h[0] = *(const v8bf*)&s_in[cur][wid * 16 + lrow][kloc];
            a.h[1] = *(const v8bf*)&s_in[cur][wid * 16 + lrow][kloc + 16];
            const int kglob = c * KC + ks * 32 + lhalf * 8;
#pragma unroll
            for (int nt = 0; nt < NT1; ++nt) {
                FragBF b;
                const __bf16* wp = Wg + (size_t)(nt * 16 + lrow) * F_DIM + kglob;
                b.h[0] = *(const v8bf*)wp;
                b.h[1] = *(const v8bf*)(wp + 16);
                acc[nt] = __builtin_amdgcn_wmma_f32_16x16x32_bf16(
                    false, a.f, false, b.f, (short)0, acc[nt], false, false);
            }
        }
        if (havenext) {
#pragma unroll
            for (int i = 0; i < 8; ++i) {
                v4bf b; b.x = (__bf16)pref[i].x; b.y = (__bf16)pref[i].y;
                b.z = (__bf16)pref[i].z; b.w = (__bf16)pref[i].w;
                *(v4bf*)&s_in[cur ^ 1][sr][(sc + i) * 4] = b;
            }
        }
        __syncthreads();
    }

    // ---- spill gates (+bias) to LDS (aliases s_in; all waves past barrier)
#pragma unroll
    for (int nt = 0; nt < NT1; ++nt) {
        const int col = nt * 16 + lrow;
        const float bias = biasC[col];
#pragma unroll
        for (int r = 0; r < 8; ++r)
            s_gates[wid * 16 + r + 8 * lhalf][col] = acc[nt][r] + bias;
    }
    __syncthreads();

    // ---- pointwise LSTM cell (h_prev = 0, c_prev = 0) --------------------
    for (int idx = tid; idx < BLOCK_M * HID_SZ; idx += 256) {
        const int row = idx / HID_SZ;
        const int j = idx - row * HID_SZ;
        const float gi = s_gates[row][j];
        const float gg = s_gates[row][2 * HID_SZ + j];
        const float go = s_gates[row][3 * HID_SZ + j];
        const float cc = sigm_f(gi) * tanh_f(gg);
        const float hh = sigm_f(go) * tanh_f(cc);
        s_h[row][j] = (__bf16)hh;
    }
    for (int idx = tid; idx < BLOCK_M * (HPAD - HID_SZ); idx += 256) {
        const int row = idx / (HPAD - HID_SZ);
        const int j = idx - row * (HPAD - HID_SZ);
        s_h[row][HID_SZ + j] = (__bf16)0.0f;
    }
    __syncthreads();

    // ---- GEMM2: out[128 x 1024] = h @ W_lin^T + b_lin --------------------
    FragBF ha[2];
#pragma unroll
    for (int ks = 0; ks < 2; ++ks) {
        const int kloc = ks * 32 + lhalf * 8;
        ha[ks].h[0] = *(const v8bf*)&s_h[wid * 16 + lrow][kloc];
        ha[ks].h[1] = *(const v8bf*)&s_h[wid * 16 + lrow][kloc + 16];
    }
    for (int nt = 0; nt < F_DIM / 16; ++nt) {
        v8f a2 = vz;
#pragma unroll
        for (int ks = 0; ks < 2; ++ks) {
            FragBF b;
            const __bf16* wp =
                Wl + (size_t)(nt * 16 + lrow) * HPAD + ks * 32 + lhalf * 8;
            b.h[0] = *(const v8bf*)wp;
            b.h[1] = *(const v8bf*)(wp + 16);
            a2 = __builtin_amdgcn_wmma_f32_16x16x32_bf16(
                false, ha[ks].f, false, b.f, (short)0, a2, false, false);
        }
        const int col = nt * 16 + lrow;
        const float bl = b_lin[col];
#pragma unroll
        for (int r = 0; r < 8; ++r) {
            const size_t grow = (size_t)rowBlock + wid * 16 + r + 8 * lhalf;
            out[grow * F_DIM + col] = a2[r] + bl;
        }
    }
}

// ---------------------------------------------------------------------------
extern "C" void kernel_launch(void* const* d_in, const int* in_sizes, int n_in,
                              void* d_out, int out_size, void* d_ws, size_t ws_size,
                              hipStream_t stream) {
    const float* in    = (const float*)d_in[0];
    const float* W_ih  = (const float*)d_in[1];
    // d_in[2] = W_hh: unused (h_prev is always zero in the reference)
    const float* b_ih  = (const float*)d_in[3];
    const float* b_hh  = (const float*)d_in[4];
    const float* W_lin = (const float*)d_in[5];
    const float* b_lin = (const float*)d_in[6];

    char* ws = (char*)d_ws;
    __bf16* Wg    = (__bf16*)(ws + WG_OFF);
    __bf16* Wl    = (__bf16*)(ws + WL_OFF);
    float*  biasC = (float*)(ws + BIAS_OFF);

    lstm_prep_weights<<<(NG * F_DIM + 255) / 256, 256, 0, stream>>>(
        W_ih, b_ih, b_hh, W_lin, Wg, Wl, biasC);

    lstm_fused<<<M_TOTAL / BLOCK_M, 256, 0, stream>>>(
        in, Wg, Wl, biasC, b_lin, (float*)d_out);
}
